// SelfAttention_8847632630203
// MI455X (gfx1250) — matrix operand order repaired and checked by reading, not compiled, not run on hardware
//
#include <hip/hip_runtime.h>
#include <hip/hip_bf16.h>
#include <math.h>

// ---------------------------------------------------------------------------
// Self-attention (N=4, L=2048, E=1024, H=16, D=64) + fc_out, bf16 WMMA path.
// 64-key chunks, software-pipelined global->reg->LDS staging, cooperative
// fp32->bf16 conversion (v_perm_b32), all WMMA operands move as b128.
// ---------------------------------------------------------------------------

typedef __attribute__((ext_vector_type(16))) __bf16   v16bf;
typedef __attribute__((ext_vector_type(8)))  float    v8f;
typedef __attribute__((ext_vector_type(8)))  unsigned v8u;

#define EMBED 1024
#define HEADS 16
#define HDIM  64
#define SEQ   2048
#define NB    4
#define SM_SCALE 0.03125f   // 1/sqrt(1024)

// pack two fp32 -> packed bf16 dword (round-to-nearest) with one v_perm_b32
__device__ __forceinline__ unsigned packbf(float lo, float hi) {
  unsigned ul = __builtin_bit_cast(unsigned, lo) + 0x8000u;
  unsigned uh = __builtin_bit_cast(unsigned, hi) + 0x8000u;
  return __builtin_amdgcn_perm(uh, ul, 0x07060302u);
}

__device__ __forceinline__ v16bf asbf(v8u u) {
  return __builtin_bit_cast(v16bf, u);
}

__device__ __forceinline__ __bf16 f2bf1(float f) {
  unsigned u = __builtin_bit_cast(unsigned, f) + 0x8000u;
  unsigned short h = (unsigned short)(u >> 16);
  return __builtin_bit_cast(__bf16, h);
}

// two aligned 16B chunks -> one 32-byte WMMA operand
__device__ __forceinline__ v8u ld2(const unsigned* p0, const unsigned* p1) {
  const uint4 a = *(const uint4*)p0;
  const uint4 b = *(const uint4*)p1;
  v8u r;
  r[0] = a.x; r[1] = a.y; r[2] = a.z; r[3] = a.w;
  r[4] = b.x; r[5] = b.y; r[6] = b.z; r[7] = b.w;
  return r;
}

// ---------------------------------------------------------------------------
// Kernel 0: w_out fp32 -> packed bf16 (row-major 1024x1024), once.
// ---------------------------------------------------------------------------
__global__ __launch_bounds__(256)
void cvt_w_kernel(const float* __restrict__ w, unsigned* __restrict__ wbf) {
  const int i = blockIdx.x * 256 + threadIdx.x;     // float4 index
  const float4 f = ((const float4*)w)[i];
  wbf[2 * i]     = packbf(f.x, f.y);
  wbf[2 * i + 1] = packbf(f.z, f.w);
}

// ---------------------------------------------------------------------------
// Kernel 1: flash attention. 8 waves/block share one (n,h); each wave owns a
// 16-query tile. 64-key chunks staged cooperatively in LDS (bf16 K and V^T),
// pipelined: next chunk loads to registers while current chunk computes.
//   S^T = K_tile(16x32d) x Q^T(32d x 16q)  (C layout: lane=query, reg=key)
//   which IS the A layout for O += P(16q x 32k) x V(32k x 16d): no shuffles.
// ---------------------------------------------------------------------------
__global__ __launch_bounds__(256)
void attn_flash_kernel(const float* __restrict__ values,
                       const float* __restrict__ keys,
                       const float* __restrict__ query,
                       __bf16* __restrict__ ws) {
  // K chunk: 64 keys x 64 dims, packed bf16 pairs along dim (+4 dword pad)
  __shared__ unsigned sK[64][36];
  // V^T chunk: 64 dims x 64 keys, packed bf16 pairs along key (+4 dword pad)
  __shared__ unsigned sVT[64][36];

  const int tid  = threadIdx.x;
  const int lane = tid & 31;
  const int wave = tid >> 5;
  const int lo   = lane & 15;
  const int hi   = lane >> 4;

  const int nh = blockIdx.x;           // n*HEADS + h
  const int n  = nh / HEADS;
  const int h  = nh - n * HEADS;
  const int q0 = (blockIdx.y * 8 + wave) * 16;

  const size_t rowStride = (size_t)HEADS * HDIM;   // 1024 floats / seq step
  const float* qbase = query  + (size_t)n * SEQ * rowStride + (size_t)h * HDIM;
  const float* kbase = keys   + (size_t)n * SEQ * rowStride + (size_t)h * HDIM;
  const float* vbase = values + (size_t)n * SEQ * rowStride + (size_t)h * HDIM;

  // ---- Q as B-operand (Q^T): lane = query column, K rows contiguous
  v8u qB[2];
  {
    const float* qr = qbase + (size_t)(q0 + lo) * rowStride;
#pragma unroll
    for (int c = 0; c < 2; ++c) {
      const float* p = qr + c * 32 + hi * 16;
#pragma unroll
      for (int r = 0; r < 8; ++r)
        qB[c][r] = packbf(p[2 * r], p[2 * r + 1]);
    }
  }

  // cooperative-staging map (256 threads, 64-key chunk)
  const int krow = tid >> 2;            // 0..63 : key row
  const int kdim = (tid & 3) * 16;      // dim start: 0,16,32,48
  const int vdim = tid & 63;            // 0..63 : dim row of V^T
  const int vkey = (tid >> 6) * 16;     // key start: 0,16,32,48

  // next-chunk staging registers (software pipeline)
  float4 ka0, ka1, ka2, ka3;            // 16 contiguous K floats
  float  va[16];                        // 16 strided V floats (one dim)

  auto loadChunk = [&](int kt) {
    const float* kg = kbase + (size_t)(kt + krow) * rowStride + kdim;
    ka0 = ((const float4*)kg)[0];
    ka1 = ((const float4*)kg)[1];
    ka2 = ((const float4*)kg)[2];
    ka3 = ((const float4*)kg)[3];
    __builtin_prefetch(kg + 64 * rowStride, 0, 3);
    const float* vg = vbase + (size_t)(kt + vkey) * rowStride + vdim;
#pragma unroll
    for (int i = 0; i < 16; ++i)
      va[i] = vg[(size_t)i * rowStride];
    __builtin_prefetch(vg + 64 * rowStride, 0, 3);
  };

  auto storeChunk = [&]() {
    uint4 p0, p1;
    p0.x = packbf(ka0.x, ka0.y); p0.y = packbf(ka0.z, ka0.w);
    p0.z = packbf(ka1.x, ka1.y); p0.w = packbf(ka1.z, ka1.w);
    p1.x = packbf(ka2.x, ka2.y); p1.y = packbf(ka2.z, ka2.w);
    p1.z = packbf(ka3.x, ka3.y); p1.w = packbf(ka3.z, ka3.w);
    *(uint4*)&sK[krow][(kdim >> 1)]     = p0;
    *(uint4*)&sK[krow][(kdim >> 1) + 4] = p1;
    uint4 q0v, q1v;
    q0v.x = packbf(va[0],  va[1]);  q0v.y = packbf(va[2],  va[3]);
    q0v.z = packbf(va[4],  va[5]);  q0v.w = packbf(va[6],  va[7]);
    q1v.x = packbf(va[8],  va[9]);  q1v.y = packbf(va[10], va[11]);
    q1v.z = packbf(va[12], va[13]); q1v.w = packbf(va[14], va[15]);
    *(uint4*)&sVT[vdim][(vkey >> 1)]     = q0v;
    *(uint4*)&sVT[vdim][(vkey >> 1) + 4] = q1v;
  };

  float m = -1e30f;     // running max (per query lo; duplicated in lane pair)
  float l = 0.0f;       // running sum
  v8f O[4] = {};        // 16q x 64d context, f32 accumulators

  loadChunk(0);

  for (int kt = 0; kt < SEQ; kt += 64) {
    storeChunk();                        // convert staged regs -> LDS
    __syncthreads();
    if (kt + 64 < SEQ) loadChunk(kt + 64);   // uniform branch; loads in flight
                                             // during the compute phase below
    // ---- S^T tiles from LDS: st[g] = keys[kt+16g .. kt+16g+15]
    v8f st[4] = {};
#pragma unroll
    for (int g = 0; g < 4; ++g) {
#pragma unroll
      for (int c = 0; c < 2; ++c) {
        const int off = c * 16 + 8 * hi;     // dwords: K = c*32 + 16*hi
        const v8u kA = ld2(&sK[g * 16 + lo][off], &sK[g * 16 + lo][off + 4]);
        st[g] = __builtin_amdgcn_wmma_f32_16x16x32_bf16(false, asbf(kA), false,
                    asbf(qB[c]), (short)0, st[g], false, false);
      }
    }

    // ---- online softmax over this 64-key chunk (per query = lane&15)
    float tmax = -1e30f;
#pragma unroll
    for (int g = 0; g < 4; ++g)
#pragma unroll
      for (int r = 0; r < 8; ++r)
        tmax = fmaxf(tmax, st[g][r]);
    tmax *= SM_SCALE;
    tmax = fmaxf(tmax, __shfl_xor(tmax, 16, 32));
    const float mn    = fmaxf(m, tmax);
    const float alpha = __expf(m - mn);

    float e[4][8];
    float rs = 0.0f;
#pragma unroll
    for (int g = 0; g < 4; ++g)
#pragma unroll
      for (int r = 0; r < 8; ++r) {
        e[g][r] = __expf(st[g][r] * SM_SCALE - mn);
        rs += e[g][r];
      }
    // P in A layout: pA0 = keys 0..31 (tiles 0,1), pA1 = keys 32..63 (2,3)
    v8u pA0, pA1;
#pragma unroll
    for (int r = 0; r < 4; ++r) {
      pA0[r]     = packbf(e[0][2 * r], e[0][2 * r + 1]);
      pA0[r + 4] = packbf(e[1][2 * r], e[1][2 * r + 1]);
      pA1[r]     = packbf(e[2][2 * r], e[2][2 * r + 1]);
      pA1[r + 4] = packbf(e[3][2 * r], e[3][2 * r + 1]);
    }
    rs += __shfl_xor(rs, 16, 32);
    l = l * alpha + rs;
    m = mn;

    // ---- rescale O: register r holds query (r + 8*hi); alpha lives in lane q
#pragma unroll
    for (int r = 0; r < 8; ++r) {
      const float a = __shfl(alpha, r + (hi << 3), 32);
      O[0][r] *= a; O[1][r] *= a; O[2][r] *= a; O[3][r] *= a;
    }

    // ---- O += P x V from LDS V^T: lane = out-dim column, contiguous keys
#pragma unroll
    for (int t = 0; t < 4; ++t) {
      const unsigned* vrow = &sVT[t * 16 + lo][0];
      const v8u vB0 = ld2(vrow + hi * 4,      vrow + 8 + hi * 4);
      O[t] = __builtin_amdgcn_wmma_f32_16x16x32_bf16(false, asbf(pA0), false,
                 asbf(vB0), (short)0, O[t], false, false);
      const v8u vB1 = ld2(vrow + 16 + hi * 4, vrow + 24 + hi * 4);
      O[t] = __builtin_amdgcn_wmma_f32_16x16x32_bf16(false, asbf(pA1), false,
                 asbf(vB1), (short)0, O[t], false, false);
    }
    __syncthreads();                     // LDS reads done before next store
  }

  // ---- normalize by row sum and store bf16 context to ws (N*L, 1024)
#pragma unroll
  for (int r = 0; r < 8; ++r) {
    const float lq  = __shfl(l, r + (hi << 3), 32);
    const float inv = 1.0f / lq;
    __bf16* wr = ws + (size_t)((size_t)n * SEQ + q0 + r + 8 * hi) * EMBED
                    + h * HDIM + lo;
#pragma unroll
    for (int t = 0; t < 4; ++t)
      wr[t * 16] = f2bf1(O[t][r] * inv);
  }
}

// ---------------------------------------------------------------------------
// Kernel 2: out = ctx @ w_out^T + b_out.  ctx: (8192,1024) bf16, wu: bf16
// weights (1024,1024) row-major. Pure b128 operand loads + WMMA.
// ---------------------------------------------------------------------------
__global__ __launch_bounds__(256)
void fc_out_kernel(const unsigned* __restrict__ xu,   // bf16 pairs, 512/row
                   const unsigned* __restrict__ wu,   // bf16 pairs, 512/row
                   const float* __restrict__ bias,
                   float* __restrict__ out) {
  const int lane = threadIdx.x & 31;
  const int wave = threadIdx.x >> 5;
  const int lo   = lane & 15;
  const int hi   = lane >> 4;

  const int m0   = blockIdx.x * 16;                 // row tile
  const int nCol = (blockIdx.y * 8 + wave) * 16;    // column tile

  v8f acc = {};
  const unsigned* xr = xu + (size_t)(m0 + lo) * (EMBED / 2);    // A: lane = row
  const unsigned* wr = wu + (size_t)(nCol + lo) * (EMBED / 2);  // B: lane = col

  for (int k0 = 0; k0 < EMBED; k0 += 32) {
    const int kd = k0 >> 1;
    // A dwords: lo half at K=k0+8*hi, hi half at K=k0+16+8*hi
    const v8u a = ld2(xr + kd + 4 * hi, xr + kd + 8 + 4 * hi);
    // B dwords: lo half at K=k0+8*hi, hi half at K=k0+16+8*hi
    const v8u b = ld2(wr + kd + 4 * hi, wr + kd + 8 + 4 * hi);
    acc = __builtin_amdgcn_wmma_f32_16x16x32_bf16(false, asbf(a), false,
              asbf(b), (short)0, acc, false, false);
  }

  const float bv = bias[nCol + lo];
#pragma unroll
  for (int r = 0; r < 8; ++r)
    out[(size_t)(m0 + r + 8 * hi) * EMBED + nCol + lo] = acc[r] + bv;
}

// ---------------------------------------------------------------------------
extern "C" void kernel_launch(void* const* d_in, const int* in_sizes, int n_in,
                              void* d_out, int out_size, void* d_ws, size_t ws_size,
                              hipStream_t stream) {
  (void)in_sizes; (void)n_in; (void)out_size; (void)ws_size;
  const float* values = (const float*)d_in[0];
  const float* keys   = (const float*)d_in[1];
  const float* query  = (const float*)d_in[2];
  const float* w_out  = (const float*)d_in[3];
  const float* b_out  = (const float*)d_in[4];
  float* out = (float*)d_out;

  // ws layout: [0,16MB) bf16 context (8192x1024); [16MB,18MB) bf16 w_out
  unsigned* wsu = (unsigned*)d_ws;
  __bf16*   ctx = (__bf16*)wsu;
  unsigned* wbf = wsu + (size_t)(NB * SEQ) * (EMBED / 2);

  // Kernel 0: convert w_out once (1024 blocks x 256 threads x 1 float4)
  hipLaunchKernelGGL(cvt_w_kernel, dim3((EMBED * EMBED / 4) / 256), dim3(256),
                     0, stream, w_out, wbf);

  // Kernel 1: 64 (n,h) blocks x 16 q-tile groups, 8 waves/block
  dim3 grdA(NB * HEADS, SEQ / (16 * 8));
  hipLaunchKernelGGL(attn_flash_kernel, grdA, dim3(256), 0, stream,
                     values, keys, query, ctx);

  // Kernel 2: 512 row tiles x 8 groups of 8 column tiles
  dim3 grdB((NB * SEQ) / 16, (EMBED / 16) / 8);
  hipLaunchKernelGGL(fc_out_kernel, grdB, dim3(256), 0, stream,
                     (const unsigned*)ctx, wbf, b_out, out);
}
